// INT4Linear_11570641895704
// MI455X (gfx1250) — matrix operand order repaired and checked by reading, not compiled, not run on hardware
//
#include <hip/hip_runtime.h>

// ---------------------------------------------------------------------------
// INT4 Linear (per-tensor scale) as f16 WMMA GEMM for gfx1250 (MI455X).
//   out[M,N] = x[M,K] @ (q[N,K]*scale)^T + bias[N]
//   M = 4*2048 = 8192, N = 16384, K = 4096
// HBM-bound problem (~0.9 GB moved vs ~1.1 TFLOP) -> minimize bytes:
//   - weights converted int32->f16 on the fly (int4 values exact in f16)
//   - x converted f32->f16 on the fly (x fits in 192MB L2 for cross-block reuse)
//   - fp32 accumulation inside V_WMMA_F32_16X16X32_F16
// ---------------------------------------------------------------------------

typedef _Float16 v16h __attribute__((ext_vector_type(16)));
typedef _Float16 v8h  __attribute__((ext_vector_type(8)));
typedef _Float16 v4h  __attribute__((ext_vector_type(4)));
typedef float    v8f  __attribute__((ext_vector_type(8)));

#define IN_F   4096
#define OUT_F  16384
#define MROWS  8192          // B*S = 4*2048

#define BM 128               // block tile M
#define BN 256               // block tile N
#define BK 32                // K step (= WMMA K)
#define KSTEPS (IN_F / BK)   // 128

__global__ __launch_bounds__(256)
void int4linear_wmma_kernel(const float* __restrict__ X,
                            const int*   __restrict__ W,
                            const float* __restrict__ scale,
                            const float* __restrict__ bias,
                            float*       __restrict__ Out)
{
    // Double-buffered f16 staging tiles: A = x slice [BM][BK], B = w slice [BN][BK]
    __shared__ __align__(16) _Float16 As[2][BM * BK];   // 2 * 8 KB
    __shared__ __align__(16) _Float16 Bs[2][BN * BK];   // 2 * 16 KB

    const int tid  = threadIdx.x;        // 0..255
    const int wave = tid >> 5;           // 0..7
    const int lane = tid & 31;
    const int hf   = lane >> 4;          // lane half (0/1) per WMMA VGPR layout
    const int lr   = lane & 15;
    const int wm   = wave >> 2;          // 0..1  (wave row in 2x4 wave grid)
    const int wn   = wave & 3;           // 0..3  (wave col)

    const int m0 = blockIdx.y * BM;
    const int n0 = blockIdx.x * BN;

    // ---- global->register prefetch state -----------------------------------
    // A tile: BM*BK fp32 = 1024 float4 -> 4 float4 per thread
    // B tile: BN*BK int32 = 2048 int4  -> 8 int4  per thread
    float4 pa[4];
    int4   pb[8];

    auto load_tiles = [&](int kk) {
        #pragma unroll
        for (int i = 0; i < 4; ++i) {
            int idx = tid + i * 256;
            int r = idx >> 3, c = idx & 7;          // row in tile, float4 col
            pa[i] = *(const float4*)(X + (size_t)(m0 + r) * IN_F + kk + c * 4);
        }
        #pragma unroll
        for (int i = 0; i < 8; ++i) {
            int idx = tid + i * 256;
            int r = idx >> 3, c = idx & 7;
            pb[i] = *(const int4*)(W + (size_t)(n0 + r) * IN_F + kk + c * 4);
        }
    };

    auto store_tiles = [&](int buf) {
        #pragma unroll
        for (int i = 0; i < 4; ++i) {
            int idx = tid + i * 256;
            int r = idx >> 3, c = idx & 7;
            v4h h;
            h[0] = (_Float16)pa[i].x;  h[1] = (_Float16)pa[i].y;
            h[2] = (_Float16)pa[i].z;  h[3] = (_Float16)pa[i].w;
            *(v4h*)&As[buf][r * BK + c * 4] = h;    // ds_store_b64
        }
        #pragma unroll
        for (int i = 0; i < 8; ++i) {
            int idx = tid + i * 256;
            int r = idx >> 3, c = idx & 7;
            v4h h;
            h[0] = (_Float16)pb[i].x;  h[1] = (_Float16)pb[i].y;
            h[2] = (_Float16)pb[i].z;  h[3] = (_Float16)pb[i].w;
            *(v4h*)&Bs[buf][r * BK + c * 4] = h;    // ds_store_b64
        }
    };

    // 4x4 grid of 16x16 f32 accumulators per wave (64x64 of C)
    v8f acc[4][4] = {};

    // ---- prologue: fill buffer 0 ------------------------------------------
    load_tiles(0);
    store_tiles(0);
    __syncthreads();

    // ---- main K loop -------------------------------------------------------
    for (int kt = 0; kt < KSTEPS; ++kt) {
        const int cur = kt & 1;

        // issue next slice's global loads early (overlaps with WMMA below)
        if (kt + 1 < KSTEPS) load_tiles((kt + 1) * BK);

        const _Float16* Ab = As[cur];
        const _Float16* Bb = Bs[cur];

        // B fragments: 32x16 f16; lane = column (n), lane-half owns K 0-15/16-31
        v16h bfrag[4];
        #pragma unroll
        for (int nt = 0; nt < 4; ++nt) {
            int col = wn * 64 + nt * 16 + lr;
            v8h blo = *(const v8h*)&Bb[col * BK + hf * 16];
            v8h bhi = *(const v8h*)&Bb[col * BK + hf * 16 + 8];
            bfrag[nt] = __builtin_shufflevector(blo, bhi,
                0,1,2,3,4,5,6,7,8,9,10,11,12,13,14,15);
        }

        #pragma unroll
        for (int mt = 0; mt < 4; ++mt) {
            // A fragment: 16x32 f16; lane = row (m), lane-half owns K 0-15/16-31
            int row = wm * 64 + mt * 16 + lr;
            v8h alo = *(const v8h*)&Ab[row * BK + hf * 16];
            v8h ahi = *(const v8h*)&Ab[row * BK + hf * 16 + 8];
            v16h afrag = __builtin_shufflevector(alo, ahi,
                0,1,2,3,4,5,6,7,8,9,10,11,12,13,14,15);

            #pragma unroll
            for (int nt = 0; nt < 4; ++nt) {
                acc[mt][nt] = __builtin_amdgcn_wmma_f32_16x16x32_f16(
                    /*neg_a=*/false, afrag,
                    /*neg_b=*/false, bfrag[nt],
                    /*c_mod=*/(short)0, acc[mt][nt],
                    /*reuse_a=*/false, /*reuse_b=*/false);
            }
        }

        // stage next slice into the other buffer (safe: all waves passed the
        // previous barrier, so nobody still reads it)
        if (kt + 1 < KSTEPS) store_tiles((kt + 1) & 1);
        __syncthreads();
    }

    // ---- epilogue: out = acc*scale + bias ---------------------------------
    // C/D layout: VGPR r, lanes 0-15 -> M=r, lanes 16-31 -> M=8+r; N = lane&15
    const float sc = scale[0];
    #pragma unroll
    for (int nt = 0; nt < 4; ++nt) {
        int n = n0 + wn * 64 + nt * 16 + lr;
        float bv = bias[n];
        #pragma unroll
        for (int mt = 0; mt < 4; ++mt) {
            int mbase = m0 + wm * 64 + mt * 16 + hf * 8;
            #pragma unroll
            for (int r = 0; r < 8; ++r) {
                Out[(size_t)(mbase + r) * OUT_F + n] = acc[mt][nt][r] * sc + bv;
            }
        }
    }
}

extern "C" void kernel_launch(void* const* d_in, const int* in_sizes, int n_in,
                              void* d_out, int out_size, void* d_ws, size_t ws_size,
                              hipStream_t stream)
{
    const float* x     = (const float*)d_in[0];
    const int*   w     = (const int*)  d_in[1];
    const float* scale = (const float*)d_in[2];
    const float* bias  = (const float*)d_in[3];
    float*       out   = (float*)d_out;

    dim3 grid(OUT_F / BN, MROWS / BM);   // 64 x 64 workgroups
    int4linear_wmma_kernel<<<grid, 256, 0, stream>>>(x, w, scale, bias, out);
}
